// DeformableQSA_32615981646079
// MI455X (gfx1250) — compile-verified, matrix operand order checked by reading, not run on hardware
//
#include <hip/hip_runtime.h>
#include <hip/hip_bf16.h>
#include <cstdint>

// ---- problem constants (match reference) ----
#define B_   4
#define M_   8
#define H_   8
#define P_   4
#define C_   256
#define HD_  32
#define NQ_  1280
#define NS_  1280

typedef __bf16 bf16_t;
typedef __attribute__((ext_vector_type(16))) __bf16 v16bf;
typedef __attribute__((ext_vector_type(8)))  __bf16 v8bf;
typedef __attribute__((ext_vector_type(8)))  float  v8f;
typedef __attribute__((ext_vector_type(4)))  float  v4f;
typedef __attribute__((ext_vector_type(4)))  int    v4i;

#if defined(__has_builtin)
#if __has_builtin(__builtin_amdgcn_global_load_async_to_lds_b128) && \
    __has_builtin(__builtin_amdgcn_s_wait_asynccnt)
#define USE_ASYNC_LDS 1
#endif
#endif

// ---------------------------------------------------------------------------
// f32 -> bf16 elementwise convert
// ---------------------------------------------------------------------------
__global__ void cvt_f32_bf16(const float* __restrict__ in,
                             bf16_t* __restrict__ out, int n) {
  int i = blockIdx.x * blockDim.x + threadIdx.x;
  if (i < n) out[i] = (bf16_t)in[i];
}

// ---------------------------------------------------------------------------
// W[K,N] f32 row-major -> Wt[N,K] bf16 row-major (B-fragment K contiguous)
// ---------------------------------------------------------------------------
__global__ void transpose_f32_bf16(const float* __restrict__ W,
                                   bf16_t* __restrict__ Wt, int K, int N) {
  int i = blockIdx.x * blockDim.x + threadIdx.x;
  if (i < K * N) {
    int k = i / N, n = i - k * N;
    Wt[n * K + k] = (bf16_t)W[i];
  }
}

// ---------------------------------------------------------------------------
// C[M,N] (f32) = A[M,K] bf16 rm  x  Wt[N,K]^T bf16 rm  + bias
//
// Block = 256 threads = 8 waves -> 128-row x 64-col macro-tile.
// Double-buffered LDS staging of the 64x32 B K-chunk via
// global_load_async_to_lds_b128: iteration k issues the async fill of
// buffer k+1 BEFORE computing from buffer k, and drains ASYNCcnt only at the
// loop-tail barrier, overlapping DMA with the 4 WMMAs per wave.
// Requires Mrows % 128 == 0, N % 64 == 0, K % 32 == 0.
// ---------------------------------------------------------------------------
__global__ void __launch_bounds__(256)
wmma_gemm_bf16(const bf16_t* __restrict__ A,
               const bf16_t* __restrict__ Wt,
               const float* __restrict__ bias,
               float* __restrict__ C,
               int Mrows, int N, int K) {
  __shared__ bf16_t Bs[2][64 * 32];              // double-buffered B K-chunks

  const int tid  = threadIdx.x;
  const int lane = tid & 31;
  const int wave = tid >> 5;                     // 0..7
  const int stripsN = N >> 6;
  const int tn = blockIdx.x % stripsN;           // 64-col strip
  const int tm = blockIdx.x / stripsN;           // 128-row block

  const int fr = lane & 15;                      // row (A) / col (B) in tile
  const int kg = (lane >> 4) << 3;               // 0 or 8 (K sub-group)

  const bf16_t* __restrict__ Arow =
      A + (size_t)(tm * 128 + wave * 16 + fr) * K;

  // cooperative B staging: thread -> (row = tid/4, k-offset = (tid%4)*8)
  const int brow = tid >> 2;                     // 0..63
  const int bk8  = (tid & 3) << 3;               // 0,8,16,24
  const bf16_t* __restrict__ Bsrc =
      Wt + (size_t)(tn * 64 + brow) * K + bk8;
  const int bdoff = brow * 32 + bk8;

  v8f acc0 = {}, acc1 = {}, acc2 = {}, acc3 = {};
  const int NK = K >> 5;

  // prologue: stage chunk 0 into buffer 0
#if USE_ASYNC_LDS
  __builtin_amdgcn_global_load_async_to_lds_b128(
      (__attribute__((address_space(1))) v4i*)Bsrc,
      (__attribute__((address_space(3))) v4i*)&Bs[0][bdoff], 0, 0);
  __builtin_amdgcn_s_wait_asynccnt(0);
#else
  *reinterpret_cast<v8bf*>(&Bs[0][bdoff]) = *reinterpret_cast<const v8bf*>(Bsrc);
#endif
  __syncthreads();

  for (int kc = 0; kc < NK; ++kc) {
    const int cur = kc & 1;
    const int k0  = kc << 5;

    // issue async fill of the NEXT chunk before computing (overlap DMA)
    if (kc + 1 < NK) {
#if USE_ASYNC_LDS
      __builtin_amdgcn_global_load_async_to_lds_b128(
          (__attribute__((address_space(1))) v4i*)(Bsrc + (k0 + 32)),
          (__attribute__((address_space(3))) v4i*)&Bs[cur ^ 1][bdoff], 0, 0);
#else
      *reinterpret_cast<v8bf*>(&Bs[cur ^ 1][bdoff]) =
          *reinterpret_cast<const v8bf*>(Bsrc + (k0 + 32));
#endif
    }

    // A fragment: two contiguous 8x bf16 chunks (k = kg..kg+7, 16+kg..16+kg+7)
    v16bf af;
    {
      v8bf lo = *reinterpret_cast<const v8bf*>(Arow + k0 + kg);
      v8bf hi = *reinterpret_cast<const v8bf*>(Arow + k0 + 16 + kg);
#pragma unroll
      for (int i = 0; i < 8; ++i) { af[i] = lo[i]; af[8 + i] = hi[i]; }
    }

    // 4 B fragments from LDS buffer `cur`, 4 WMMAs reusing af
#pragma unroll
    for (int c = 0; c < 4; ++c) {
      const bf16_t* bp = &Bs[cur][(c * 16 + fr) * 32];
      v16bf bf;
      {
        v8bf lo = *reinterpret_cast<const v8bf*>(bp + kg);
        v8bf hi = *reinterpret_cast<const v8bf*>(bp + 16 + kg);
#pragma unroll
        for (int i = 0; i < 8; ++i) { bf[i] = lo[i]; bf[8 + i] = hi[i]; }
      }
      if (c == 0)
        acc0 = __builtin_amdgcn_wmma_f32_16x16x32_bf16(false, af, false, bf,
                                                       (short)0, acc0, false, false);
      else if (c == 1)
        acc1 = __builtin_amdgcn_wmma_f32_16x16x32_bf16(false, af, false, bf,
                                                       (short)0, acc1, false, false);
      else if (c == 2)
        acc2 = __builtin_amdgcn_wmma_f32_16x16x32_bf16(false, af, false, bf,
                                                       (short)0, acc2, false, false);
      else
        acc3 = __builtin_amdgcn_wmma_f32_16x16x32_bf16(false, af, false, bf,
                                                       (short)0, acc3, false, false);
    }

    // drain the in-flight DMA and make the next buffer visible to all waves
#if USE_ASYNC_LDS
    __builtin_amdgcn_s_wait_asynccnt(0);
#endif
    __syncthreads();
  }

  // epilogue: lane holds column (lane&15); VGPR r -> row r (+8 for hi lanes)
  const int rbase = tm * 128 + wave * 16 + ((lane >> 4) ? 8 : 0);
#pragma unroll
  for (int c = 0; c < 4; ++c) {
    const int cn = tn * 64 + c * 16 + fr;
    const float bv = bias ? bias[cn] : 0.0f;
    const v8f acc = (c == 0) ? acc0 : (c == 1) ? acc1 : (c == 2) ? acc2 : acc3;
#pragma unroll
    for (int r = 0; r < 8; ++r)
      C[(size_t)(rbase + r) * N + cn] = acc[r] + bv;
  }
}

// ---------------------------------------------------------------------------
// s_mean[m][c] = mean_n s_feat[m][n][c]
// ---------------------------------------------------------------------------
__global__ void smean_kernel(const float* __restrict__ s_feat,
                             float* __restrict__ smean) {
  int i = blockIdx.x * blockDim.x + threadIdx.x;       // M_*C_
  if (i >= M_ * C_) return;
  int m = i >> 8, c = i & (C_ - 1);
  float s = 0.f;
  for (int n = 0; n < NS_; ++n) s += s_feat[((size_t)m * NS_ + n) * C_ + c];
  smean[i] = s * (1.0f / NS_);
}

// ---------------------------------------------------------------------------
// d_s[m][j] = bd[j] + sum_c smean[m][c] * Wd[C_ + c][j]     (j < 64)
// ---------------------------------------------------------------------------
__global__ void ds_kernel(const float* __restrict__ smean,
                          const float* __restrict__ Wd,
                          const float* __restrict__ bd,
                          float* __restrict__ ds) {
  int i = blockIdx.x * blockDim.x + threadIdx.x;       // M_*64
  if (i >= M_ * 64) return;
  int m = i >> 6, j = i & 63;
  float s = bd[j];
  for (int c = 0; c < C_; ++c)
    s += smean[m * C_ + c] * Wd[(size_t)(C_ + c) * 64 + j];
  ds[i] = s;
}

// ---------------------------------------------------------------------------
// Deformable gather + 4-point attention.  One thread per (b,m,h,q).
// Writes att[(b,m,q), h*32+d] as bf16 (feeds the final WMMA GEMM).
// ---------------------------------------------------------------------------
__global__ void attn_kernel(const float* __restrict__ q,
                            const float* __restrict__ k,
                            const float* __restrict__ v,
                            const float* __restrict__ dq,
                            const float* __restrict__ ds,
                            const long long* __restrict__ q_shapes,
                            const long long* __restrict__ s_shapes,
                            bf16_t* __restrict__ att) {
  int t = blockIdx.x * blockDim.x + threadIdx.x;
  if (t >= B_ * M_ * H_ * NQ_) return;
  const int qi = t % NQ_;  int r = t / NQ_;
  const int h  = r % H_;   r /= H_;
  const int m  = r % M_;   const int b = r / M_;

  const int q0sq = (int)(q_shapes[0] * q_shapes[0]);
  const int lvl = (qi < q0sq) ? 0 : 1;
  const float spl = (float)s_shapes[lvl];
  const float spl_sq = spl * spl;

  const float* __restrict__ dqp = dq + ((size_t)b * NQ_ + qi) * 64 + h * (2 * P_);
  const float* __restrict__ dsp = ds + m * 64 + h * (2 * P_);
  const float* __restrict__ qp  = q  + ((size_t)b * NQ_ + qi) * C_ + h * HD_;

  float qv[HD_];
  {
    const v4f* q4 = reinterpret_cast<const v4f*>(qp);
#pragma unroll
    for (int i = 0; i < HD_ / 4; ++i) {
      v4f x = q4[i];
      qv[4 * i] = x.x; qv[4 * i + 1] = x.y; qv[4 * i + 2] = x.z; qv[4 * i + 3] = x.w;
    }
  }

  float sc[P_];
  int   idxs[P_];
#pragma unroll
  for (int p = 0; p < P_; ++p) {
    const float a0 = dqp[2 * p + 0] + dsp[2 * p + 0];
    const float a1 = dqp[2 * p + 1] + dsp[2 * p + 1];
    const float d0 = 1.0f / (1.0f + expf(-a0));
    const float d1 = 1.0f / (1.0f + expf(-a1));
    int idx = (int)(d1 * spl + d0 * spl_sq);         // trunc, like astype(int32)
    idx = idx < 0 ? 0 : (idx > NS_ - 1 ? NS_ - 1 : idx);
    idxs[p] = idx;
    const v4f* kp = reinterpret_cast<const v4f*>(
        k + ((size_t)m * NS_ + idx) * C_ + h * HD_);
    float dot = 0.f;
#pragma unroll
    for (int i = 0; i < HD_ / 4; ++i) {
      v4f x = kp[i];
      dot += qv[4 * i] * x.x + qv[4 * i + 1] * x.y +
             qv[4 * i + 2] * x.z + qv[4 * i + 3] * x.w;
    }
    sc[p] = dot;
  }

  float mx = sc[0];
#pragma unroll
  for (int p = 1; p < P_; ++p) mx = sc[p] > mx ? sc[p] : mx;
  float sum = 0.f;
#pragma unroll
  for (int p = 0; p < P_; ++p) { sc[p] = expf(sc[p] - mx); sum += sc[p]; }
  const float inv = 1.0f / sum;

  float out[HD_];
#pragma unroll
  for (int i = 0; i < HD_; ++i) out[i] = 0.f;
#pragma unroll
  for (int p = 0; p < P_; ++p) {
    const float w = sc[p] * inv;
    const v4f* vp = reinterpret_cast<const v4f*>(
        v + ((size_t)m * NS_ + idxs[p]) * C_ + h * HD_);
#pragma unroll
    for (int i = 0; i < HD_ / 4; ++i) {
      v4f x = vp[i];
      out[4 * i] += w * x.x; out[4 * i + 1] += w * x.y;
      out[4 * i + 2] += w * x.z; out[4 * i + 3] += w * x.w;
    }
  }

  bf16_t* __restrict__ op =
      att + (((size_t)(b * M_ + m) * NQ_ + qi) * C_ + h * HD_);
#pragma unroll
  for (int i = 0; i < HD_; ++i) op[i] = (bf16_t)out[i];
}

// ---------------------------------------------------------------------------
// host-side launch
// ---------------------------------------------------------------------------
extern "C" void kernel_launch(void* const* d_in, const int* in_sizes, int n_in,
                              void* d_out, int out_size, void* d_ws, size_t ws_size,
                              hipStream_t stream) {
  (void)in_sizes; (void)n_in; (void)out_size; (void)ws_size;
  const float*     q_feat   = (const float*)d_in[0];
  const float*     s_feat   = (const float*)d_in[1];
  const long long* q_shapes = (const long long*)d_in[2];
  const long long* s_shapes = (const long long*)d_in[3];
  const float* Wq = (const float*)d_in[4];
  const float* bq = (const float*)d_in[5];
  const float* Wk = (const float*)d_in[6];
  const float* bk = (const float*)d_in[7];
  const float* Wv = (const float*)d_in[8];
  const float* bv = (const float*)d_in[9];
  const float* Wd = (const float*)d_in[10];
  const float* bd = (const float*)d_in[11];
  const float* Wp = (const float*)d_in[12];
  const float* bp = (const float*)d_in[13];
  float* out = (float*)d_out;

  size_t off = 0;
  auto alloc = [&](size_t bytes) -> void* {
    void* p = (char*)d_ws + off;
    off += (bytes + 255) & ~(size_t)255;
    return p;
  };
  const int NQr = B_ * NQ_;          // 5120 rows
  const int NSr = M_ * NS_;          // 10240 rows
  const int NAr = B_ * M_ * NQ_;     // 40960 rows
  bf16_t* qf_bf = (bf16_t*)alloc((size_t)NQr * C_ * 2);
  bf16_t* sf_bf = (bf16_t*)alloc((size_t)NSr * C_ * 2);
  bf16_t* wq_t  = (bf16_t*)alloc((size_t)C_ * C_ * 2);
  bf16_t* wk_t  = (bf16_t*)alloc((size_t)C_ * C_ * 2);
  bf16_t* wv_t  = (bf16_t*)alloc((size_t)C_ * C_ * 2);
  bf16_t* wp_t  = (bf16_t*)alloc((size_t)C_ * C_ * 2);
  bf16_t* wd_t  = (bf16_t*)alloc((size_t)64 * C_ * 2);
  float*  qbuf  = (float*)alloc((size_t)NQr * C_ * 4);
  float*  kbuf  = (float*)alloc((size_t)NSr * C_ * 4);
  float*  vbuf  = (float*)alloc((size_t)NSr * C_ * 4);
  float*  dqbuf = (float*)alloc((size_t)NQr * 64 * 4);
  float*  smean = (float*)alloc((size_t)M_ * C_ * 4);
  float*  dsbuf = (float*)alloc((size_t)M_ * 64 * 4);
  bf16_t* att   = (bf16_t*)alloc((size_t)NAr * C_ * 2);

  const int T = 256;
  auto nblk = [](int n, int t) { return (n + t - 1) / t; };

  cvt_f32_bf16<<<nblk(NQr * C_, T), T, 0, stream>>>(q_feat, qf_bf, NQr * C_);
  cvt_f32_bf16<<<nblk(NSr * C_, T), T, 0, stream>>>(s_feat, sf_bf, NSr * C_);
  transpose_f32_bf16<<<nblk(C_ * C_, T), T, 0, stream>>>(Wq, wq_t, C_, C_);
  transpose_f32_bf16<<<nblk(C_ * C_, T), T, 0, stream>>>(Wk, wk_t, C_, C_);
  transpose_f32_bf16<<<nblk(C_ * C_, T), T, 0, stream>>>(Wv, wv_t, C_, C_);
  transpose_f32_bf16<<<nblk(C_ * C_, T), T, 0, stream>>>(Wp, wp_t, C_, C_);
  transpose_f32_bf16<<<nblk(C_ * 64, T), T, 0, stream>>>(Wd, wd_t, C_, 64);

  auto gemm = [&](const bf16_t* Ap, const bf16_t* Wtp, const float* bias,
                  float* Cp, int Mr, int N) {
    const int blocks = (Mr / 128) * (N / 64);
    wmma_gemm_bf16<<<blocks, 256, 0, stream>>>(Ap, Wtp, bias, Cp, Mr, N, C_);
  };
  gemm(qf_bf, wq_t, bq, qbuf, NQr, C_);
  gemm(sf_bf, wk_t, bk, kbuf, NSr, C_);
  gemm(sf_bf, wv_t, bv, vbuf, NSr, C_);
  gemm(qf_bf, wd_t, nullptr, dqbuf, NQr, 64);

  smean_kernel<<<nblk(M_ * C_, T), T, 0, stream>>>(s_feat, smean);
  ds_kernel<<<nblk(M_ * 64, T), T, 0, stream>>>(smean, Wd, bd, dsbuf);

  attn_kernel<<<nblk(B_ * M_ * H_ * NQ_, T), T, 0, stream>>>(
      qbuf, kbuf, vbuf, dqbuf, dsbuf, q_shapes, s_shapes, att);

  gemm(att, wp_t, bp, out, NAr, C_);
}
